// DecoderND_39058432590521
// MI455X (gfx1250) — compile-verified, matrix operand order checked by reading, not run on hardware
//
#include <hip/hip_runtime.h>
#include <hip/hip_bf16.h>

// ---------------------------------------------------------------------------
// Problem constants (from reference)
// ---------------------------------------------------------------------------
#define BV 32000
#define EDIM 512
#define HDIM 1024
#define TWO_H 2048
#define THREE_H 3072
#define BB 16
#define TT 64
#define TXX 128

typedef __bf16 bf16_t;
typedef bf16_t v16bf __attribute__((ext_vector_type(16)));
typedef bf16_t v8bf  __attribute__((ext_vector_type(8)));
typedef float  v8f   __attribute__((ext_vector_type(8)));
typedef float  v4f   __attribute__((ext_vector_type(4)));
typedef unsigned u32x4 __attribute__((ext_vector_type(4)));

// truncating f32->bf16 pack: two floats -> one dword (lo in [15:0], hi in [31:16])
__device__ __forceinline__ unsigned pack_bf16x2(float lo, float hi) {
    unsigned ul = __builtin_bit_cast(unsigned, lo);
    unsigned uh = __builtin_bit_cast(unsigned, hi);
    return (uh & 0xFFFF0000u) | (ul >> 16);
}

// load 8 consecutive f32 (16B-aligned) and convert to v8bf
__device__ __forceinline__ v8bf cvt8(const float* __restrict__ p) {
    v4f a = *(const v4f*)p;
    v4f b = *(const v4f*)(p + 4);
    u32x4 u;
    u[0] = pack_bf16x2(a[0], a[1]);
    u[1] = pack_bf16x2(a[2], a[3]);
    u[2] = pack_bf16x2(b[0], b[1]);
    u[3] = pack_bf16x2(b[2], b[3]);
    return __builtin_bit_cast(v8bf, u);
}

__device__ __forceinline__ v16bf join16(v8bf lo, v8bf hi) {
    return __builtin_shufflevector(lo, hi, 0,1,2,3,4,5,6,7,8,9,10,11,12,13,14,15);
}

__device__ __forceinline__ v8f wmma_bf16(v16bf a, v16bf b, v8f c) {
    return __builtin_amdgcn_wmma_f32_16x16x32_bf16(false, a, false, b, (short)0, c, false, false);
}

// LDS row padding: 32 payload bf16 + 8 pad = 40 (80B row -> 20-bank stride,
// conflict-free 16B reads across 16 lanes on 64 banks)
#define LROW 40

// ---------------------------------------------------------------------------
// Big tiled GEMM:  C[M,N] = A[M,K] @ W[:, k0:k0+K]^T + bias
// block = 256 threads (8 waves), tile 128x64, K stepped by 32, bf16 WMMA.
// Each wave owns a 2x2 grid of 16x16 output tiles (4 WMMA / K-step).
// grid = (N/64, M/128). M mult of 128, N mult of 64, K mult of 32; all
// leading dims multiples of 4 floats (16B).
// ---------------------------------------------------------------------------
__global__ void gemm_bf16_128x64(const float* __restrict__ A, int lda,
                                 const float* __restrict__ W, int ldw, int k0, int K,
                                 const float* __restrict__ bias,
                                 float* __restrict__ C, int ldc)
{
    __shared__ bf16_t As[128][LROW];  // As[m][k]  (k contiguous)
    __shared__ bf16_t Bs[64][LROW];   // Bs[n][k]  (k contiguous, W-transposed tile)

    const int m0  = blockIdx.y * 128;
    const int n0  = blockIdx.x * 64;
    const int tid = threadIdx.x;
    const int lane = tid & 31;
    const int wv   = tid >> 5;        // wave 0..7
    const int mp   = wv & 3;          // m-tile pair 0..3 -> tiles 2mp, 2mp+1
    const int np   = wv >> 2;         // n-tile pair 0..1 -> tiles 2np, 2np+1

    const int row   = lane & 15;
    const int kbase = (lane < 16) ? 0 : 8;

    v8f acc00 = {}, acc01 = {}, acc10 = {}, acc11 = {};

    const int sm = tid >> 2;            // 0..63
    const int sk = (tid & 3) * 8;       // 0,8,16,24

    for (int kb = 0; kb < K; kb += 32) {
        // ---- stage A tile (128x32): two 32B global reads + two 16B LDS stores ----
        #pragma unroll
        for (int i = 0; i < 2; ++i) {
            const int c  = tid + 256 * i;      // 0..511
            const int r  = c >> 2;             // 0..127
            const int ck = (c & 3) * 8;
            const float* srcA = A + (size_t)(m0 + r) * lda + kb + ck;
            *(v8bf*)&As[r][ck] = cvt8(srcA);
        }
        // ---- stage B tile transposed (64x32): W rows contiguous in k ----
        {
            const float* srcW = W + (size_t)(n0 + sm) * ldw + (size_t)k0 + kb + sk;
            *(v8bf*)&Bs[sm][sk] = cvt8(srcW);
            if (kb + 32 < K)  // prefetch next K-slab of W into near cache
                __builtin_prefetch(srcW + 32, 0, 3);
        }
        __syncthreads();

        // ---- fragments: two contiguous 16B LDS reads each (ds_load_b128) ----
        const bf16_t* ap0 = &As[(2 * mp) * 16 + row][kbase];
        const bf16_t* ap1 = &As[(2 * mp + 1) * 16 + row][kbase];
        const bf16_t* bp0 = &Bs[(2 * np) * 16 + row][kbase];
        const bf16_t* bp1 = &Bs[(2 * np + 1) * 16 + row][kbase];
        v16bf a0 = join16(*(const v8bf*)ap0, *(const v8bf*)(ap0 + 16));
        v16bf a1 = join16(*(const v8bf*)ap1, *(const v8bf*)(ap1 + 16));
        v16bf b0 = join16(*(const v8bf*)bp0, *(const v8bf*)(bp0 + 16));
        v16bf b1 = join16(*(const v8bf*)bp1, *(const v8bf*)(bp1 + 16));

        acc00 = wmma_bf16(a0, b0, acc00);
        acc01 = wmma_bf16(a0, b1, acc01);
        acc10 = wmma_bf16(a1, b0, acc10);
        acc11 = wmma_bf16(a1, b1, acc11);
        __syncthreads();
    }

    // ---- store D: lane -> col = lane&15, rows j (+8 for lanes 16..31) ----
    const int cn    = lane & 15;
    const int rbase = (lane < 16) ? 0 : 8;
    const int n1 = n0 + (2 * np) * 16 + cn;
    const int n2 = n0 + (2 * np + 1) * 16 + cn;
    const float b1v = bias ? bias[n1] : 0.f;
    const float b2v = bias ? bias[n2] : 0.f;
    #pragma unroll
    for (int j = 0; j < 8; ++j) {
        const int ma = m0 + (2 * mp) * 16 + rbase + j;
        const int mb = m0 + (2 * mp + 1) * 16 + rbase + j;
        C[(size_t)ma * ldc + n1] = acc00[j] + b1v;
        C[(size_t)ma * ldc + n2] = acc01[j] + b2v;
        C[(size_t)mb * ldc + n1] = acc10[j] + b1v;
        C[(size_t)mb * ldc + n2] = acc11[j] + b2v;
    }
}

// ---------------------------------------------------------------------------
// Skinny GEMM, M = 16 fixed:  C[16,N] = A[16,K] @ W[:, k0:k0+K]^T
// block = 256 threads (8 waves); each wave owns a 16-wide column strip.
// grid.x = N / 128.  K multiple of 32; lda/ldw/k0 multiples of 4.
// ---------------------------------------------------------------------------
__global__ void gemm_skinny16(const float* __restrict__ A, int lda,
                              const float* __restrict__ W, int ldw, int k0, int K,
                              float* __restrict__ C, int ldc)
{
    const int tid  = threadIdx.x;
    const int lane = tid & 31;
    const int wv   = tid >> 5;
    const int n0   = blockIdx.x * 128 + wv * 16;

    const int row   = lane & 15;             // A row / W output-column within tile
    const int kbase = (lane < 16) ? 0 : 8;

    const float* arow = A + (size_t)row * lda + kbase;
    const float* wrow = W + (size_t)(n0 + row) * ldw + (size_t)k0 + kbase;

    v8f acc = {};
    for (int kb = 0; kb < K; kb += 32) {
        v16bf a = join16(cvt8(arow + kb), cvt8(arow + kb + 16));
        v16bf b = join16(cvt8(wrow + kb), cvt8(wrow + kb + 16));
        acc = wmma_bf16(a, b, acc);
    }

    const int cn    = lane & 15;
    const int rbase = (lane < 16) ? 0 : 8;
    #pragma unroll
    for (int j = 0; j < 8; ++j)
        C[(size_t)(rbase + j) * ldc + (n0 + cn)] = acc[j];
}

// ---------------------------------------------------------------------------
// Fused additive attention per step: q = s@Wa^T+b; sc = tanh(q+UaK)@va;
// softmax; c = sc @ keys.  One block per batch row, 256 threads.
// ---------------------------------------------------------------------------
__global__ void attn_kernel(const float* __restrict__ s,
                            const float* __restrict__ Wa_w, const float* __restrict__ Wa_b,
                            const float* __restrict__ UaK,  const float* __restrict__ va,
                            const float* __restrict__ keys, float* __restrict__ c)
{
    const int b   = blockIdx.x;
    const int tid = threadIdx.x;

    __shared__ float sS[HDIM];
    __shared__ float q[HDIM];
    __shared__ float sc[TXX];
    __shared__ float red[2];

    for (int i = tid; i < HDIM; i += 256) sS[i] = s[b * HDIM + i];
    __syncthreads();

    // q[h] = Wa_b[h] + s . Wa_w[h,:]
    for (int h = tid; h < HDIM; h += 256) {
        const float* wr = Wa_w + (size_t)h * HDIM;
        float acc = Wa_b[h];
        for (int k = 0; k < HDIM; k += 4) {
            v4f w4 = *(const v4f*)(wr + k);
            v4f s4 = *(const v4f*)(sS + k);
            acc += s4[0]*w4[0] + s4[1]*w4[1] + s4[2]*w4[2] + s4[3]*w4[3];
        }
        q[h] = acc;
    }
    __syncthreads();

    // scores: 2 threads per tx, each covers half of H
    {
        const int tx = tid >> 1, half = tid & 1;
        const float* u  = UaK + ((size_t)b * TXX + tx) * HDIM + half * 512;
        const float* qh = q  + half * 512;
        const float* vh = va + half * 512;
        float acc = 0.f;
        for (int k = 0; k < 512; ++k) acc += tanhf(qh[k] + u[k]) * vh[k];
        if (half == 0) sc[tx] = acc;
        __syncthreads();
        if (half == 1) sc[tx] += acc;
    }
    __syncthreads();

    if (tid == 0) {
        float mx = -1e30f;
        for (int i = 0; i < TXX; ++i) mx = fmaxf(mx, sc[i]);
        red[0] = mx;
    }
    __syncthreads();
    if (tid < TXX) sc[tid] = __expf(sc[tid] - red[0]);
    __syncthreads();
    if (tid == 0) {
        float sm = 0.f;
        for (int i = 0; i < TXX; ++i) sm += sc[i];
        red[1] = 1.f / sm;
    }
    __syncthreads();

    const float inv = red[1];
    for (int k = tid; k < TWO_H; k += 256) {
        const float* kp = keys + (size_t)b * TXX * TWO_H + k;
        float acc = 0.f;
        for (int tx = 0; tx < TXX; ++tx) acc += sc[tx] * kp[(size_t)tx * TWO_H];
        c[b * TWO_H + k] = acc * inv;
    }
}

// ---------------------------------------------------------------------------
// GRU gate fusion + state/output write.
// ---------------------------------------------------------------------------
__global__ void gru_gate(const float* __restrict__ gi_x, const float* __restrict__ gc,
                         const float* __restrict__ gh,
                         const float* __restrict__ bih, const float* __restrict__ bhh,
                         float* __restrict__ h, float* __restrict__ hs, int t)
{
    const int e = blockIdx.x * blockDim.x + threadIdx.x;
    if (e >= BB * HDIM) return;
    const int b = e >> 10, j = e & (HDIM - 1);

    const size_t rg = (size_t)(b * TT + t) * THREE_H;
    const size_t rc = (size_t)b * THREE_H;

    const float ir = gi_x[rg + j]            + gc[rc + j]            + bih[j];
    const float iz = gi_x[rg + HDIM + j]     + gc[rc + HDIM + j]     + bih[HDIM + j];
    const float in = gi_x[rg + 2*HDIM + j]   + gc[rc + 2*HDIM + j]   + bih[2*HDIM + j];
    const float hr = gh[rc + j]              + bhh[j];
    const float hz = gh[rc + HDIM + j]       + bhh[HDIM + j];
    const float hn = gh[rc + 2*HDIM + j]     + bhh[2*HDIM + j];

    const float r = 1.f / (1.f + __expf(-(ir + hr)));
    const float z = 1.f / (1.f + __expf(-(iz + hz)));
    const float n = tanhf(in + r * hn);
    const float hp = h[e];
    const float hnew = (1.f - z) * n + z * hp;
    h[e] = hnew;
    hs[(size_t)(b * TT + t) * HDIM + j] = hnew;
}

// s0 = keys[:,0,H:2H] @ initialWs[n]   (no transpose)
__global__ void init_s0(const float* __restrict__ keys, const float* __restrict__ Wn,
                        float* __restrict__ h)
{
    const int e = blockIdx.x * 256 + threadIdx.x;
    if (e >= BB * HDIM) return;
    const int b = e >> 10, j = e & (HDIM - 1);
    const float* k0 = keys + (size_t)b * TXX * TWO_H + HDIM;
    float acc = 0.f;
    for (int k = 0; k < HDIM; ++k) acc += k0[k] * Wn[(size_t)k * HDIM + j];
    h[e] = acc;
}

__global__ void embed_kernel(const int* __restrict__ x_t, const float* __restrict__ emb,
                             float* __restrict__ x0)
{
    const int i = blockIdx.x * 256 + threadIdx.x;
    if (i >= BB * TT * EDIM) return;
    const int e = i & (EDIM - 1);
    const int row = i >> 9;
    x0[i] = emb[(size_t)x_t[row] * EDIM + e];
}

__global__ void skip_add(const float* __restrict__ a, const float* __restrict__ b,
                         float* __restrict__ o, int n)
{
    const int i = blockIdx.x * 256 + threadIdx.x;
    if (i < n) o[i] = a[i] + b[i];
}

// ---------------------------------------------------------------------------
// Host orchestration
// ---------------------------------------------------------------------------
extern "C" void kernel_launch(void* const* d_in, const int* in_sizes, int n_in,
                              void* d_out, int out_size, void* d_ws, size_t ws_size,
                              hipStream_t stream)
{
    const int*   x_t   = (const int*)d_in[0];
    const float* enc   = (const float*)d_in[1];
    const float* emb   = (const float*)d_in[2];
    const float* Wa_w  = (const float*)d_in[3];
    const float* Wa_b  = (const float*)d_in[4];
    const float* Ua_w  = (const float*)d_in[5];
    const float* Ua_b  = (const float*)d_in[6];
    const float* Va_w  = (const float*)d_in[7];
    const float* Wih[2] = {(const float*)d_in[8],  (const float*)d_in[12]};
    const float* Whh[2] = {(const float*)d_in[9],  (const float*)d_in[13]};
    const float* bih[2] = {(const float*)d_in[10], (const float*)d_in[14]};
    const float* bhh[2] = {(const float*)d_in[11], (const float*)d_in[15]};
    const float* out_w = (const float*)d_in[16];
    const float* out_b = (const float*)d_in[17];
    const float* initW = (const float*)d_in[18];

    float* ws   = (float*)d_ws;
    float* x0   = ws;                   // [B*T, E]            524288
    float* UaK  = x0   + 524288;        // [B*Tx, H]          2097152
    float* gi_x = UaK  + 2097152;       // [B*T, 3H]          3145728
    float* h    = gi_x + 3145728;       // [B, H]               16384
    float* cbuf = h    + 16384;         // [B, 2H]              32768
    float* gc   = cbuf + 32768;         // [B, 3H]              49152
    float* gh   = gc   + 49152;         // [B, 3H]              49152
    float* hs0  = gh   + 49152;         // [B*T, H]           1048576
    float* hs1  = hs0  + 1048576;       // [B*T, H]           1048576
    float* skip = hs1  + 1048576;       // [B*T, H]           1048576

    // 1) embedding gather
    embed_kernel<<<(BB * TT * EDIM + 255) / 256, 256, 0, stream>>>(x_t, emb, x0);

    for (int n = 0; n < 2; ++n) {
        const float* keys = enc + (size_t)n * BB * TXX * TWO_H;
        float* hsN = (n == 0) ? hs0 : hs1;

        // 2a) UaK = keys @ Ua_w^T + Ua_b    (M=2048, N=1024, K=2048)
        gemm_bf16_128x64<<<dim3(HDIM / 64, (BB * TXX) / 128), 256, 0, stream>>>(
            keys, TWO_H, Ua_w, TWO_H, 0, TWO_H, Ua_b, UaK, HDIM);

        // 2b) s0 = keys[:,0,H:] @ initialWs[n]
        init_s0<<<(BB * HDIM) / 256, 256, 0, stream>>>(
            keys, initW + (size_t)n * HDIM * HDIM, h);

        // 2c) hoisted input-path gates: gi_x = X @ Wih[:, :Din]^T (all timesteps)
        const float* Xn = (n == 0) ? x0 : hs0;
        const int Din   = (n == 0) ? EDIM : HDIM;
        const int ldw   = (n == 0) ? (EDIM + TWO_H) : THREE_H;
        gemm_bf16_128x64<<<dim3(THREE_H / 64, (BB * TT) / 128), 256, 0, stream>>>(
            Xn, Din, Wih[n], ldw, 0, Din, nullptr, gi_x, THREE_H);

        // 2d) sequential scan over T
        for (int t = 0; t < TT; ++t) {
            attn_kernel<<<BB, 256, 0, stream>>>(h, Wa_w, Wa_b, UaK, Va_w, keys, cbuf);
            // gc = c @ Wih[:, Din:Din+2H]^T
            gemm_skinny16<<<THREE_H / 128, 256, 0, stream>>>(
                cbuf, TWO_H, Wih[n], ldw, Din, TWO_H, gc, THREE_H);
            // gh = h @ Whh^T
            gemm_skinny16<<<THREE_H / 128, 256, 0, stream>>>(
                h, HDIM, Whh[n], HDIM, 0, HDIM, gh, THREE_H);
            gru_gate<<<(BB * HDIM) / 256, 256, 0, stream>>>(
                gi_x, gc, gh, bih[n], bhh[n], h, hsN, t);
        }
    }

    // 3) skip = hs0 + hs1
    skip_add<<<(BB * TT * HDIM + 255) / 256, 256, 0, stream>>>(
        hs0, hs1, skip, BB * TT * HDIM);

    // 4) logits = skip @ out_w^T + out_b   (M=1024, N=32000, K=1024)
    gemm_bf16_128x64<<<dim3(BV / 64, (BB * TT) / 128), 256, 0, stream>>>(
        skip, HDIM, out_w, HDIM, 0, HDIM, out_b, (float*)d_out, BV);
}
